// GATModule_80470507258221
// MI455X (gfx1250) — matrix operand order repaired and checked by reading, not compiled, not run on hardware
//
#include <hip/hip_runtime.h>

typedef __attribute__((ext_vector_type(16))) _Float16 v16h;
typedef __attribute__((ext_vector_type(8)))  float    v8f;

#define CCH 64
#define HC  128
#define NEG_SLOPE 0.2f

// ---- order-preserving float <-> uint encoding (for atomicMax on floats) ----
__device__ __forceinline__ unsigned enc_f(float f) {
  unsigned u = __float_as_uint(f);
  return (u & 0x80000000u) ? ~u : (u | 0x80000000u);
}
__device__ __forceinline__ float dec_f(unsigned e) {
  return (e & 0x80000000u) ? __uint_as_float(e ^ 0x80000000u) : __uint_as_float(~e);
}
__device__ __forceinline__ float leaky(float v) {
  return v > 0.0f ? v : NEG_SLOPE * v;
}
__device__ __forceinline__ void edge_sd(const int* __restrict__ ei, int i, int E,
                                        int& s, int& d) {
  if (i < E) { s = ei[i]; d = ei[E + i]; }
  else       { s = d = i - E; }            // appended self-loops
}

// ---- h = x @ W  (fp32 in, f16 WMMA, fp32 accumulate), h: [N, 128] ----------
// One wave computes one 16x16 tile of h. blockDim=256 (8 waves), wave w owns
// column tile n0=16*w (128/16=8 column tiles), blockIdx.x owns row tile.
__global__ void gat_gemm(const float* __restrict__ x, const float* __restrict__ W,
                         float* __restrict__ h, int Nn) {
  const int lane = threadIdx.x & 31;
  const int wave = threadIdx.x >> 5;
  const int m0 = blockIdx.x * 16;
  const int n0 = wave * 16;
  const int mrow = lane & 15;            // A row within tile
  const int ncol = lane & 15;            // B/D column within tile
  const int hk   = lane >> 4;            // lane half selector

  int rowA = m0 + mrow; if (rowA >= Nn) rowA = Nn - 1;   // clamp (EXEC stays full for WMMA)

  v8f c = {};
#pragma unroll
  for (int kc = 0; kc < 2; ++kc) {
    const int ka = kc * 32;
    v16h a, b;
    const float* xrow = x + (size_t)rowA * CCH + ka;
#pragma unroll
    for (int v = 0; v < 8; ++v) {
      // A fragment: contiguous K order, VGPR v -> K=hk*16+2v{,+1} (matches B)
      int baseA = hk * 16 + 2 * v;
      float2 xv = *(const float2*)(xrow + baseA);
      a[2 * v]     = (_Float16)xv.x;
      a[2 * v + 1] = (_Float16)xv.y;
      // B fragment: lanes 0-15 hold K=0..15, lanes 16-31 K=16..31; VGPR v -> K=hk*16+2v{,+1}
      int baseB = hk * 16 + 2 * v;
      const float* wp = W + (size_t)(ka + baseB) * HC + n0 + ncol;
      b[2 * v]     = (_Float16)wp[0];
      b[2 * v + 1] = (_Float16)wp[HC];
    }
    c = __builtin_amdgcn_wmma_f32_16x16x32_f16(false, a, false, b, (short)0, c,
                                               false, false);
  }
  // D: VGPR r -> row m = (lane>>4)*8 + r, col = lane&15
  const int mbase = m0 + hk * 8;
  float* hp = h + (size_t)mbase * HC + n0 + ncol;
  if (m0 + 16 <= Nn) {             // uniform: whole tile in range -> straight stores
#pragma unroll
    for (int r = 0; r < 8; ++r) hp[(size_t)r * HC] = c[r];
  } else {                         // ragged tail tile only
#pragma unroll
    for (int r = 0; r < 8; ++r)
      if (mbase + r < Nn) hp[(size_t)r * HC] = c[r];
  }
}

// ---- per-node attention logits: al[n][head] = sum_c h[n,head,c]*a[head,c] --
__global__ void gat_al(const float* __restrict__ h, const float* __restrict__ as,
                       const float* __restrict__ ad, float* __restrict__ al_s,
                       float* __restrict__ al_d, int Nn) {
  const int node = (blockIdx.x * blockDim.x + threadIdx.x) >> 5;
  const int lane = threadIdx.x & 31;
  if (node >= Nn) return;
  const float* hp = h + (size_t)node * HC;
  float h0a = hp[lane], h0b = hp[lane + 32];       // head 0 channels
  float h1a = hp[lane + 64], h1b = hp[lane + 96];  // head 1 channels
  float s0 = h0a * as[lane] + h0b * as[lane + 32];
  float s1 = h1a * as[lane + 64] + h1b * as[lane + 96];
  float d0 = h0a * ad[lane] + h0b * ad[lane + 32];
  float d1 = h1a * ad[lane + 64] + h1b * ad[lane + 96];
#pragma unroll
  for (int off = 16; off > 0; off >>= 1) {
    s0 += __shfl_xor(s0, off, 32);
    s1 += __shfl_xor(s1, off, 32);
    d0 += __shfl_xor(d0, off, 32);
    d1 += __shfl_xor(d1, off, 32);
  }
  if (lane == 0) {
    al_s[2 * node] = s0; al_s[2 * node + 1] = s1;
    al_d[2 * node] = d0; al_d[2 * node + 1] = d1;
  }
}

// ---- pass 1: segment max over dst (uint-encoded float atomicMax) -----------
__global__ void gat_edge_max(const int* __restrict__ ei, int E, int ET,
                             const float* __restrict__ al_s,
                             const float* __restrict__ al_d,
                             unsigned* __restrict__ m_enc) {
  int i = blockIdx.x * blockDim.x + threadIdx.x;
  if (i >= ET) return;
  int s, d; edge_sd(ei, i, E, s, d);
  float2 sl = *(const float2*)(al_s + 2 * s);
  float2 dl = *(const float2*)(al_d + 2 * d);
  atomicMax(m_enc + 2 * d,     enc_f(leaky(sl.x + dl.x)));
  atomicMax(m_enc + 2 * d + 1, enc_f(leaky(sl.y + dl.y)));
}

// ---- pass 2: segment sum of exp(e - m) -------------------------------------
__global__ void gat_edge_sum(const int* __restrict__ ei, int E, int ET,
                             const float* __restrict__ al_s,
                             const float* __restrict__ al_d,
                             const unsigned* __restrict__ m_enc,
                             float* __restrict__ denom) {
  int i = blockIdx.x * blockDim.x + threadIdx.x;
  if (i >= ET) return;
  int s, d; edge_sd(ei, i, E, s, d);
  float2 sl = *(const float2*)(al_s + 2 * s);
  float2 dl = *(const float2*)(al_d + 2 * d);
  float e0 = leaky(sl.x + dl.x), e1 = leaky(sl.y + dl.y);
  atomicAdd(denom + 2 * d,     __expf(e0 - dec_f(m_enc[2 * d])));
  atomicAdd(denom + 2 * d + 1, __expf(e1 - dec_f(m_enc[2 * d + 1])));
}

// ---- pass 3: weighted scatter-add; one wave per edge, float4 per lane ------
__global__ void gat_edge_agg(const int* __restrict__ ei, int E, int ET,
                             const float* __restrict__ al_s,
                             const float* __restrict__ al_d,
                             const unsigned* __restrict__ m_enc,
                             const float* __restrict__ denom,
                             const float* __restrict__ h,
                             float* __restrict__ agg) {
  const int i    = (blockIdx.x * blockDim.x + threadIdx.x) >> 5;
  const int lane = threadIdx.x & 31;
  if (i >= ET) return;
  int s, d; edge_sd(ei, i, E, s, d);
  float2 sl = *(const float2*)(al_s + 2 * s);
  float2 dl = *(const float2*)(al_d + 2 * d);
  float e0 = leaky(sl.x + dl.x), e1 = leaky(sl.y + dl.y);
  float a0 = __expf(e0 - dec_f(m_enc[2 * d])) / (denom[2 * d] + 1e-16f);
  float a1 = __expf(e1 - dec_f(m_enc[2 * d + 1])) / (denom[2 * d + 1] + 1e-16f);
  float alpha = (lane < 16) ? a0 : a1;   // channels 4*lane..4*lane+3 share a head
  const float4 hv = *(const float4*)(h + (size_t)s * HC + lane * 4);
  float* ap = agg + (size_t)d * HC + lane * 4;
  atomicAdd(ap + 0, hv.x * alpha);
  atomicAdd(ap + 1, hv.y * alpha);
  atomicAdd(ap + 2, hv.z * alpha);
  atomicAdd(ap + 3, hv.w * alpha);
}

// ---- head mean + bias + residual, float4-vectorized (in-place safe) --------
__global__ void gat_finalize(const float* __restrict__ agg,
                             const float* __restrict__ bias,
                             const float* __restrict__ xin,
                             float* __restrict__ xout, int Nn) {
  int idx = blockIdx.x * blockDim.x + threadIdx.x;     // one float4 per thread
  if (idx >= Nn * (CCH / 4)) return;
  int n = idx >> 4, c4 = (idx & 15) * 4;
  const float4 g0 = *(const float4*)(agg + (size_t)n * HC + c4);
  const float4 g1 = *(const float4*)(agg + (size_t)n * HC + 64 + c4);
  const float4 bv = *(const float4*)(bias + c4);
  const float4 xv = *(const float4*)(xin + (size_t)n * CCH + c4);
  float4 o;
  o.x = 0.5f * (g0.x + g1.x) + bv.x + xv.x;
  o.y = 0.5f * (g0.y + g1.y) + bv.y + xv.y;
  o.z = 0.5f * (g0.z + g1.z) + bv.z + xv.z;
  o.w = 0.5f * (g0.w + g1.w) + bv.w + xv.w;
  *(float4*)(xout + (size_t)n * CCH + c4) = o;
}

extern "C" void kernel_launch(void* const* d_in, const int* in_sizes, int n_in,
                              void* d_out, int out_size, void* d_ws, size_t ws_size,
                              hipStream_t stream) {
  const float* x0 = (const float*)d_in[0];
  const int*   ei = (const int*)d_in[1];
  const int Nn = in_sizes[0] / CCH;
  const int E  = in_sizes[1] / 2;
  const int ET = E + Nn;

  float* ws    = (float*)d_ws;
  float* xa    = ws;                          // Nn*64
  float* hbuf  = xa   + (size_t)Nn * CCH;     // Nn*128
  float* agg   = hbuf + (size_t)Nn * HC;      // Nn*128
  float* al_s  = agg  + (size_t)Nn * HC;      // Nn*2
  float* al_d  = al_s + (size_t)Nn * 2;       // Nn*2
  float* denom = al_d + (size_t)Nn * 2;       // Nn*2
  unsigned* m_enc = (unsigned*)(denom + (size_t)Nn * 2);  // Nn*2

  const float* xin = x0;
  for (int l = 0; l < 3; ++l) {
    const float* W  = (const float*)d_in[2 + 4 * l];
    const float* as = (const float*)d_in[3 + 4 * l];
    const float* ad = (const float*)d_in[4 + 4 * l];
    const float* b  = (const float*)d_in[5 + 4 * l];
    float* xout = (l == 0) ? xa : (float*)d_out;  // l1 temp, l2 final (in-place safe)

    hipMemsetAsync(m_enc, 0, (size_t)Nn * 2 * sizeof(unsigned), stream);
    hipMemsetAsync(denom, 0, (size_t)Nn * 2 * sizeof(float), stream);
    hipMemsetAsync(agg,   0, (size_t)Nn * HC * sizeof(float), stream);

    gat_gemm<<<dim3((Nn + 15) / 16), dim3(256), 0, stream>>>(xin, W, hbuf, Nn);
    gat_al<<<dim3((Nn + 7) / 8), dim3(256), 0, stream>>>(hbuf, as, ad, al_s, al_d, Nn);
    gat_edge_max<<<dim3((ET + 255) / 256), dim3(256), 0, stream>>>(ei, E, ET, al_s, al_d, m_enc);
    gat_edge_sum<<<dim3((ET + 255) / 256), dim3(256), 0, stream>>>(ei, E, ET, al_s, al_d, m_enc, denom);
    gat_edge_agg<<<dim3((ET + 7) / 8), dim3(256), 0, stream>>>(ei, E, ET, al_s, al_d, m_enc, denom, hbuf, agg);
    gat_finalize<<<dim3((Nn * (CCH / 4) + 255) / 256), dim3(256), 0, stream>>>(agg, b, xin, xout, Nn);
    xin = xout;
  }
}
